// QuantizerEMA_43026982372001
// MI455X (gfx1250) — matrix operand order repaired and checked by reading, not compiled, not run on hardware
//
#include <hip/hip_runtime.h>
#include <hip/hip_bf16.h>

// Problem sizes (fixed by the reference)
#define NZ 65536
#define DD 128
#define KK 2048
#define DH 64

typedef __attribute__((ext_vector_type(16))) __bf16 v16bf;
typedef __attribute__((ext_vector_type(8)))  float  v8f;

__device__ __forceinline__ void split_bf16(float x, __bf16& hi, __bf16& lo) {
    __bf16 h = (__bf16)x;
    hi = h;
    lo = (__bf16)(x - (float)h);
}

// ---------------------------------------------------------------------------
// Kernel 1: project codebook: e_ = emb @ W^T + b  -> bf16 hi/lo + exact ||e_||^2
// grid = KK blocks of DH threads (thread j -> output dim j of code k)
// ---------------------------------------------------------------------------
__global__ void k_proj_emb(const float* __restrict__ emb,
                           const float* __restrict__ W,
                           const float* __restrict__ bias,
                           __bf16* __restrict__ eh,
                           __bf16* __restrict__ el,
                           float* __restrict__ en) {
    const int k = blockIdx.x;
    const int j = threadIdx.x;              // 0..63
    const float* er = emb + (size_t)k * DD;
    const float* wr = W   + (size_t)j * DD;
    float acc = bias[j];
    #pragma unroll 8
    for (int d = 0; d < DD; ++d) acc += er[d] * wr[d];

    __bf16 hi, lo;
    split_bf16(acc, hi, lo);
    eh[(size_t)k * DH + j] = hi;
    el[(size_t)k * DH + j] = lo;

    __shared__ float red[DH];
    red[j] = acc * acc;
    __syncthreads();
    #pragma unroll
    for (int s = DH / 2; s > 0; s >>= 1) {
        if (j < s) red[j] += red[j + s];
        __syncthreads();
    }
    if (j == 0) en[k] = red[0];
}

// ---------------------------------------------------------------------------
// Kernel 2: project z with WMMA (bf16x3 emulation of fp32 GEMM):
//   zp[N,64] = z[N,128] @ W^T[128,64] + b
// block = 256 (8 waves); each wave computes a 16-row x 64-col tile.
// grid = NZ/128.
// ---------------------------------------------------------------------------
__global__ void k_proj_z(const float* __restrict__ z,
                         const float* __restrict__ W,
                         const float* __restrict__ bias,
                         float* __restrict__ zp) {
    const int wave = threadIdx.x >> 5;
    const int lane = threadIdx.x & 31;
    const int l15  = lane & 15;
    const int hv   = lane >> 4;             // lane half (0/1)
    const int m0   = blockIdx.x * 128 + wave * 16;

    const float* zrow = z + (size_t)(m0 + l15) * DD;

    v8f acc[4] = {};

    #pragma unroll
    for (int kc = 0; kc < 4; ++kc) {        // K chunks of 32 over DD=128
        // A fragment (16x32 bf16): per lane K = kc*32 + hv*16 + [0,16)
        v16bf ahi, alo;
        const int kb = kc * 32 + hv * 16;
        #pragma unroll
        for (int i = 0; i < 8; ++i) {
            __bf16 h, l;
            split_bf16(zrow[kb + i], h, l);
            ahi[i] = h; alo[i] = l;
            split_bf16(zrow[kb + 8 + i], h, l);
            ahi[8 + i] = h; alo[8 + i] = l;
        }
        #pragma unroll
        for (int nt = 0; nt < 4; ++nt) {    // 4 tiles of 16 output dims
            const int n = nt * 16 + l15;    // output dim == W row (B = W^T)
            const float* wrow = W + (size_t)n * DD + kc * 32 + hv * 16;
            v16bf bhi, blo;
            #pragma unroll
            for (int i = 0; i < 16; ++i) {
                __bf16 h, l;
                split_bf16(wrow[i], h, l);
                bhi[i] = h; blo[i] = l;
            }
            acc[nt] = __builtin_amdgcn_wmma_f32_16x16x32_bf16(
                false, ahi, false, bhi, (short)0, acc[nt], false, false);
            acc[nt] = __builtin_amdgcn_wmma_f32_16x16x32_bf16(
                false, ahi, false, blo, (short)0, acc[nt], false, false);
            acc[nt] = __builtin_amdgcn_wmma_f32_16x16x32_bf16(
                false, alo, false, bhi, (short)0, acc[nt], false, false);
        }
    }

    #pragma unroll
    for (int nt = 0; nt < 4; ++nt) {
        const float bb = bias[nt * 16 + l15];
        #pragma unroll
        for (int r = 0; r < 8; ++r) {
            // D layout: M = r + 8*hv, N = l15
            zp[(size_t)(m0 + r + 8 * hv) * DH + nt * 16 + l15] = acc[nt][r] + bb;
        }
    }
}

// ---------------------------------------------------------------------------
// Kernel 3: streaming distance argmin.
//   proxy(m,n) = ||e_n||^2 - 2 * <z_m, e_n>   (||z_m||^2 dropped: row-constant)
// Each wave owns 16 rows, streams all 128 codebook tiles of 16 codes with
// bf16x3 WMMA, keeps running (min,argmin) in registers, shuffle-reduces.
// block = 256 (8 waves), grid = NZ/128.
// ---------------------------------------------------------------------------
__global__ void k_argmin(const float* __restrict__ zp,
                         const __bf16* __restrict__ eh,
                         const __bf16* __restrict__ el,
                         const float* __restrict__ en,
                         int* __restrict__ closest) {
    const int wave = threadIdx.x >> 5;
    const int lane = threadIdx.x & 31;
    const int l15  = lane & 15;
    const int hv   = lane >> 4;
    const int m0   = blockIdx.x * 128 + wave * 16;

    // Load + split A fragments once (16 rows x 64 dims -> 2 K-chunks of 32)
    v16bf ahi[2], alo[2];
    const float* zr = zp + (size_t)(m0 + l15) * DH;
    #pragma unroll
    for (int kc = 0; kc < 2; ++kc) {
        const int kb = kc * 32 + hv * 16;
        #pragma unroll
        for (int i = 0; i < 8; ++i) {
            __bf16 h, l;
            split_bf16(zr[kb + i], h, l);
            ahi[kc][i] = h; alo[kc][i] = l;
            split_bf16(zr[kb + 8 + i], h, l);
            ahi[kc][8 + i] = h; alo[kc][8 + i] = l;
        }
    }

    float best[8];
    int   bidx[8];
    #pragma unroll
    for (int r = 0; r < 8; ++r) { best[r] = 3.4e38f; bidx[r] = 0; }

    for (int t = 0; t < KK / 16; ++t) {
        const int n = t * 16 + l15;         // this lane's codebook column
        v8f acc = {};
        #pragma unroll
        for (int kc = 0; kc < 2; ++kc) {
            // B fragment (32x16): per lane K = kc*32 + hv*16 + [0,16)
            const size_t off = (size_t)n * DH + kc * 32 + hv * 16; // 32B aligned
            v16bf bh = *(const v16bf*)(eh + off);
            v16bf bl = *(const v16bf*)(el + off);
            acc = __builtin_amdgcn_wmma_f32_16x16x32_bf16(
                false, ahi[kc], false, bh, (short)0, acc, false, false);
            acc = __builtin_amdgcn_wmma_f32_16x16x32_bf16(
                false, ahi[kc], false, bl, (short)0, acc, false, false);
            acc = __builtin_amdgcn_wmma_f32_16x16x32_bf16(
                false, alo[kc], false, bh, (short)0, acc, false, false);
        }
        const float e2 = en[n];
        #pragma unroll
        for (int r = 0; r < 8; ++r) {
            const float v = e2 - 2.0f * acc[r];
            if (v < best[r]) { best[r] = v; bidx[r] = n; }   // first-hit keeps lowest n
        }
    }

    // min+argmin reduction across the 16 lanes of each half (xor masks stay in-half)
    #pragma unroll
    for (int off = 1; off < 16; off <<= 1) {
        #pragma unroll
        for (int r = 0; r < 8; ++r) {
            const float ov = __shfl_xor(best[r], off, 32);
            const int   oi = __shfl_xor(bidx[r], off, 32);
            if (ov < best[r] || (ov == best[r] && oi < bidx[r])) {
                best[r] = ov; bidx[r] = oi;
            }
        }
    }
    if (l15 == 0) {
        #pragma unroll
        for (int r = 0; r < 8; ++r)
            closest[m0 + r + 8 * hv] = bidx[r];
    }
}

// ---------------------------------------------------------------------------
// Kernel 4: emit quantized = embeddings[closest] and the dense one-hot.
// One block (256 threads) per row; float4 stores; full rewrite every call.
// ---------------------------------------------------------------------------
__global__ void k_write(const int* __restrict__ closest,
                        const float* __restrict__ emb,
                        float* __restrict__ outq,
                        float* __restrict__ outh) {
    const int row = blockIdx.x;
    const int t   = threadIdx.x;
    const int c   = closest[row];

    if (t < DD / 4) {
        const float4* src = (const float4*)(emb + (size_t)c * DD);
        float4* q = (float4*)(outq + (size_t)row * DD);
        q[t] = src[t];
    }

    float* hrow = outh + (size_t)row * KK;
    #pragma unroll
    for (int g = 0; g < 2; ++g) {
        const int base = (t + g * 256) * 4;
        float4 v = make_float4(0.f, 0.f, 0.f, 0.f);
        if (c >= base && c < base + 4) ((float*)&v)[c - base] = 1.0f;
        *(float4*)(hrow + base) = v;
    }
}

// ---------------------------------------------------------------------------
extern "C" void kernel_launch(void* const* d_in, const int* in_sizes, int n_in,
                              void* d_out, int out_size, void* d_ws, size_t ws_size,
                              hipStream_t stream) {
    (void)in_sizes; (void)n_in; (void)out_size; (void)ws_size;

    const float* z    = (const float*)d_in[0];   // [N, D]
    const float* W    = (const float*)d_in[1];   // [DH, D]
    const float* bias = (const float*)d_in[2];   // [DH]
    const float* emb  = (const float*)d_in[3];   // [K, D]

    float* outq = (float*)d_out;                       // [N, D]
    float* outh = (float*)d_out + (size_t)NZ * DD;     // [N, K]

    char* ws = (char*)d_ws;
    size_t off = 0;
    float*  zp = (float*)(ws + off);  off += (size_t)NZ * DH * sizeof(float);   // 16 MB
    __bf16* eh = (__bf16*)(ws + off); off += (size_t)KK * DH * sizeof(__bf16);  // 256 KB
    __bf16* el = (__bf16*)(ws + off); off += (size_t)KK * DH * sizeof(__bf16);  // 256 KB
    float*  en = (float*)(ws + off);  off += (size_t)KK * sizeof(float);        // 8 KB
    int* closest = (int*)(ws + off);  off += (size_t)NZ * sizeof(int);          // 256 KB

    k_proj_emb<<<KK, DH, 0, stream>>>(emb, W, bias, eh, el, en);
    k_proj_z  <<<NZ / 128, 256, 0, stream>>>(z, W, bias, zp);
    k_argmin  <<<NZ / 128, 256, 0, stream>>>(zp, eh, el, en, closest);
    k_write   <<<NZ, 256, 0, stream>>>(closest, emb, outq, outh);
}
